// FusedCrossEntropyLMHead_17927193493831
// MI455X (gfx1250) — matrix operand order repaired and checked by reading, not compiled, not run on hardware
//
#include <hip/hip_runtime.h>

// ---------------------------------------------------------------------------
// Fused cross-entropy LM head for MI455X (gfx1250, wave32, WMMA).
//   N=8192 rows, D=2048, V=32000, label smoothing eps=0.1.
//
// Roofline: 2*N*V*D = 1.07 TFLOP  (~215us at ~5 PFLOPS dense bf16 WMMA)
//           vs 320MB HBM (~14us)  -> compute/L2 bound.
// Therefore: (1) pre-convert W and H to bf16 in workspace once (halves the
// 32GB of L2 W-streaming and removes all converts from the hot loop),
// (2) 8 WMMAs per K-chunk per wave (4 M-frags x 2 N-frags, A-frags reused),
// (3) branchless online-softmax folding 2 columns per update (1.5 exp/logit),
// (4) H tile staged into LDS with gfx1250 async global->LDS DMA.
// The 1GB logits tensor is never materialized.
// ---------------------------------------------------------------------------

typedef __attribute__((ext_vector_type(16))) __bf16 v16bf;
typedef __attribute__((ext_vector_type(8)))  __bf16 v8bf;
typedef __attribute__((ext_vector_type(4)))  __bf16 v4bf;
typedef __attribute__((ext_vector_type(8)))  float  v8f;
typedef int ce_v4i __attribute__((vector_size(4 * sizeof(int))));

#define N_ROWS   8192
#define DIM      2048
#define VOCAB    32000
#define SPLITS   5
#define VSPLIT   6400      // VOCAB / SPLITS
#define NTILE    256       // columns per V-iteration (8 waves x 2 frags x 16)
#define MTILE    64        // rows per block (4 WMMA m-fragments)
#define KCHUNKS  64        // DIM / 32
#define LDS_STRIDE 2056    // 2048 + 8 bf16 pad: rows land on distinct banks
#define NEG_BIG  (-1.0e30f)

// gfx1250 async global->LDS path (guarded: falls back to plain LDS copies)
#if defined(__has_builtin)
#if __has_builtin(__builtin_amdgcn_global_load_async_to_lds_b128)
#define CE_ASYNC_LDS 1
#endif
#endif

__device__ inline v16bf bfrag_from_bf16(const __bf16* p) {
    v8bf lo = *(const v8bf*)(p);
    v8bf hi = *(const v8bf*)(p + 8);
    return __builtin_shufflevector(lo, hi, 0,1,2,3,4,5,6,7,8,9,10,11,12,13,14,15);
}

__device__ inline v16bf bfrag_from_f32(const float* p) {
    float4 b0 = *(const float4*)(p + 0);
    float4 b1 = *(const float4*)(p + 4);
    float4 b2 = *(const float4*)(p + 8);
    float4 b3 = *(const float4*)(p + 12);
    v16bf bv;
    bv[0]=(__bf16)b0.x; bv[1]=(__bf16)b0.y; bv[2]=(__bf16)b0.z; bv[3]=(__bf16)b0.w;
    bv[4]=(__bf16)b1.x; bv[5]=(__bf16)b1.y; bv[6]=(__bf16)b1.z; bv[7]=(__bf16)b1.w;
    bv[8]=(__bf16)b2.x; bv[9]=(__bf16)b2.y; bv[10]=(__bf16)b2.z; bv[11]=(__bf16)b2.w;
    bv[12]=(__bf16)b3.x; bv[13]=(__bf16)b3.y; bv[14]=(__bf16)b3.z; bv[15]=(__bf16)b3.w;
    return bv;
}

// fp32 -> bf16 pre-conversion (8 elements / thread)
__global__ __launch_bounds__(256)
void cvt_to_bf16(const float* __restrict__ src, __bf16* __restrict__ dst, int n8)
{
    int i = blockIdx.x * 256 + threadIdx.x;
    if (i >= n8) return;
    const float4* s = (const float4*)src + (size_t)i * 2;
    float4 a = s[0], b = s[1];
    v8bf o;
    o[0]=(__bf16)a.x; o[1]=(__bf16)a.y; o[2]=(__bf16)a.z; o[3]=(__bf16)a.w;
    o[4]=(__bf16)b.x; o[5]=(__bf16)b.y; o[6]=(__bf16)b.z; o[7]=(__bf16)b.w;
    *(v8bf*)(dst + (size_t)i * 8) = o;
}

// PRECONV=true: Hsrc/Wsrc are bf16 (workspace copies); else fp32 originals.
template<bool PRECONV>
__global__ __launch_bounds__(256, 1)
void ce_gemm_stats(const void* __restrict__ Hsrc, const void* __restrict__ Wsrc,
                   float* __restrict__ pmax, float* __restrict__ psum,
                   float* __restrict__ ptot)
{
    extern __shared__ __bf16 smemH[];              // [MTILE][LDS_STRIDE]
    __shared__ float smemStats[8 * MTILE * 3];     // per-wave per-row partials

    const int tid  = threadIdx.x;
    const int lane = tid & 31;
    const int wave = tid >> 5;                      // 0..7
    const int rb   = blockIdx.x;                    // row block (0..127)
    const int sp   = blockIdx.y;                    // vocab split (0..4)
    const int row0 = rb * MTILE;

    // ---- Stage H tile (64 x 2048) into LDS as bf16 ------------------------
    if constexpr (PRECONV) {
        const __bf16* hb = (const __bf16*)Hsrc;
        for (int row = 0; row < MTILE; ++row) {
            const __bf16* src = hb + (size_t)(row0 + row) * DIM + tid * 8;
            __bf16* dst = smemH + row * LDS_STRIDE + tid * 8;
#ifdef CE_ASYNC_LDS
            __builtin_amdgcn_global_load_async_to_lds_b128(
                (__attribute__((address_space(1))) ce_v4i*)src,
                (__attribute__((address_space(3))) ce_v4i*)dst, 0, 0);
#else
            *(v8bf*)dst = *(const v8bf*)src;
#endif
        }
#ifdef CE_ASYNC_LDS
#if __has_builtin(__builtin_amdgcn_s_wait_asynccnt)
        __builtin_amdgcn_s_wait_asynccnt(0);
#else
        asm volatile("s_wait_asynccnt 0" ::: "memory");
#endif
#endif
    } else {
        const float* hf = (const float*)Hsrc;
        for (int it = 0; it < 128; ++it) {
            int slot = it * 256 + tid;              // 0..32767 float4 slots
            int r = slot >> 9;
            int c = (slot & 511) * 4;
            float4 f = *(const float4*)(hf + (size_t)(row0 + r) * DIM + c);
            v4bf hv;
            hv[0]=(__bf16)f.x; hv[1]=(__bf16)f.y; hv[2]=(__bf16)f.z; hv[3]=(__bf16)f.w;
            *(v4bf*)(smemH + r * LDS_STRIDE + c) = hv;
        }
    }
    __syncthreads();

    // ---- Per-lane online stats --------------------------------------------
    float m[4][8], s[4][8], t[4][8];
#pragma unroll
    for (int mf = 0; mf < 4; ++mf)
#pragma unroll
        for (int r = 0; r < 8; ++r) { m[mf][r] = NEG_BIG; s[mf][r] = 0.f; t[mf][r] = 0.f; }

    const int colInLane = lane & 15;                // B-matrix column N
    const int kg        = lane >> 4;                // K-group (0/1)
    // Each wave owns two 16-column fragments: rows rowA and rowA+16 of W.
    const size_t rowA = (size_t)(sp * VSPLIT + wave * 32 + colInLane);
    const size_t eoff = (size_t)kg * 16;            // element offset within row

    const __bf16* wb = (const __bf16*)Wsrc;
    const float*  wf = (const float*)Wsrc;

    for (int vt = 0; vt < VSPLIT / NTILE; ++vt) {   // 25 vocab tiles
        const size_t base0 = (rowA + (size_t)vt * NTILE) * DIM + eoff;
        const size_t base1 = base0 + (size_t)16 * DIM;
        if constexpr (PRECONV)
            __builtin_prefetch(wb + base0 + (size_t)NTILE * DIM, 0, 1);
        else
            __builtin_prefetch(wf + base0 + (size_t)NTILE * DIM, 0, 1);

        v8f acc0[4], acc1[4];
#pragma unroll
        for (int mf = 0; mf < 4; ++mf) { acc0[mf] = 0.f; acc1[mf] = 0.f; }

        for (int kc = 0; kc < KCHUNKS; ++kc) {
            v16bf bv0, bv1;
            if constexpr (PRECONV) {
                bv0 = bfrag_from_bf16(wb + base0 + kc * 32);
                bv1 = bfrag_from_bf16(wb + base1 + kc * 32);
            } else {
                bv0 = bfrag_from_f32(wf + base0 + kc * 32);
                bv1 = bfrag_from_f32(wf + base1 + kc * 32);
            }
#pragma unroll
            for (int mf = 0; mf < 4; ++mf) {
                const __bf16* ap = smemH + (mf * 16 + colInLane) * LDS_STRIDE
                                         + kc * 32 + kg * 16;
                v8bf alo = *(const v8bf*)(ap);        // K = kg*16 .. +8
                v8bf ahi = *(const v8bf*)(ap + 8);    // K = kg*16 + 8 .. +8
                v16bf av = __builtin_shufflevector(alo, ahi,
                    0,1,2,3,4,5,6,7,8,9,10,11,12,13,14,15);
                acc0[mf] = __builtin_amdgcn_wmma_f32_16x16x32_bf16(
                    false, av, false, bv0, (short)0, acc0[mf], false, false);
                acc1[mf] = __builtin_amdgcn_wmma_f32_16x16x32_bf16(
                    false, av, false, bv1, (short)0, acc1[mf], false, false);
            }
        }

        // Branchless online softmax: fold both columns, 3 exps / 2 logits.
#pragma unroll
        for (int mf = 0; mf < 4; ++mf)
#pragma unroll
            for (int r = 0; r < 8; ++r) {
                float x0 = acc0[mf][r], x1 = acc1[mf][r];
                t[mf][r] += x0 + x1;
                float nm = fmaxf(m[mf][r], fmaxf(x0, x1));
                s[mf][r] = s[mf][r] * __expf(m[mf][r] - nm)
                         + __expf(x0 - nm) + __expf(x1 - nm);
                m[mf][r] = nm;
            }
    }

    // ---- Merge across the 16 lanes that share each row --------------------
#pragma unroll
    for (int mask = 1; mask < 16; mask <<= 1) {
#pragma unroll
        for (int mf = 0; mf < 4; ++mf)
#pragma unroll
            for (int r = 0; r < 8; ++r) {
                float om = __shfl_xor(m[mf][r], mask, 32);
                float os = __shfl_xor(s[mf][r], mask, 32);
                float ot = __shfl_xor(t[mf][r], mask, 32);
                float nm = fmaxf(m[mf][r], om);
                s[mf][r] = s[mf][r] * __expf(m[mf][r] - nm) + os * __expf(om - nm);
                m[mf][r] = nm;
                t[mf][r] += ot;
            }
    }

    if (colInLane == 0) {
#pragma unroll
        for (int mf = 0; mf < 4; ++mf)
#pragma unroll
            for (int r = 0; r < 8; ++r) {
                int row  = mf * 16 + r + 8 * kg;
                int base = (wave * MTILE + row) * 3;
                smemStats[base + 0] = m[mf][r];
                smemStats[base + 1] = s[mf][r];
                smemStats[base + 2] = t[mf][r];
            }
    }
    __syncthreads();

    // ---- Merge the 8 waves, write per-(row, split) partials ---------------
    if (tid < MTILE) {
        float M = NEG_BIG, S = 0.f, T = 0.f;
#pragma unroll
        for (int w = 0; w < 8; ++w) {
            int base = (w * MTILE + tid) * 3;
            float om = smemStats[base + 0];
            float os = smemStats[base + 1];
            float ot = smemStats[base + 2];
            float nm = fmaxf(M, om);
            S = S * __expf(M - nm) + os * __expf(om - nm);
            M = nm;
            T += ot;
        }
        size_t p = (size_t)(row0 + tid) * SPLITS + sp;
        pmax[p] = M; psum[p] = S; ptot[p] = T;
    }
}

// One wave per row: z = dot(H[i], W[label_i]); merge split partials -> loss_i.
__global__ __launch_bounds__(256)
void ce_row_loss(const float* __restrict__ H, const float* __restrict__ W,
                 const int* __restrict__ labels,
                 const float* __restrict__ pmax, const float* __restrict__ psum,
                 const float* __restrict__ ptot, float* __restrict__ rowloss)
{
    const int lane = threadIdx.x & 31;
    const int wave = threadIdx.x >> 5;
    const int row  = blockIdx.x * 8 + wave;

    const int y = labels[row];
    const float* h = H + (size_t)row * DIM;
    const float* w = W + (size_t)y * DIM;

    float acc = 0.f;
#pragma unroll
    for (int j = 0; j < DIM / 128; ++j) {
        int k = (lane + j * 32) * 4;
        float4 a = *(const float4*)(h + k);
        float4 b = *(const float4*)(w + k);
        acc += a.x * b.x + a.y * b.y + a.z * b.z + a.w * b.w;
    }
#pragma unroll
    for (int mask = 16; mask >= 1; mask >>= 1)
        acc += __shfl_xor(acc, mask, 32);

    if (lane == 0) {
        float M = NEG_BIG, S = 0.f, T = 0.f;
#pragma unroll
        for (int sp = 0; sp < SPLITS; ++sp) {
            size_t p = (size_t)row * SPLITS + sp;
            float om = pmax[p], os = psum[p], ot = ptot[p];
            float nm = fmaxf(M, om);
            S = S * __expf(M - nm) + os * __expf(om - nm);
            M = nm;
            T += ot;
        }
        float lse = M + __logf(S);
        rowloss[row] = lse - 0.9f * acc - (0.1f / (float)VOCAB) * T;
    }
}

__global__ __launch_bounds__(256)
void ce_reduce(const float* __restrict__ rowloss, float* __restrict__ out)
{
    __shared__ float sm[256];
    float a = 0.f;
    for (int i = threadIdx.x; i < N_ROWS; i += 256) a += rowloss[i];
    sm[threadIdx.x] = a;
    __syncthreads();
    for (int step = 128; step > 0; step >>= 1) {
        if (threadIdx.x < step) sm[threadIdx.x] += sm[threadIdx.x + step];
        __syncthreads();
    }
    if (threadIdx.x == 0) out[0] = sm[0] * (1.0f / (float)N_ROWS);
}

extern "C" void kernel_launch(void* const* d_in, const int* in_sizes, int n_in,
                              void* d_out, int out_size, void* d_ws, size_t ws_size,
                              hipStream_t stream) {
    const float* H      = (const float*)d_in[0];   // (B,T,D) fp32
    const int*   labels = (const int*)d_in[1];     // (B,T)
    const float* W      = (const float*)d_in[2];   // (V,D) fp32
    float* out = (float*)d_out;

    const size_t wElems = (size_t)VOCAB * DIM;     // 65,536,000
    const size_t hElems = (size_t)N_ROWS * DIM;    // 16,777,216
    const size_t wBytes = wElems * 2;
    const size_t hBytes = hElems * 2;
    const size_t statBytes = (size_t)N_ROWS * SPLITS * 3 * 4 + (size_t)N_ROWS * 4;
    const size_t needed = wBytes + hBytes + statBytes;

    const bool preconv = (ws_size >= needed);
    const int ldsBytes = MTILE * LDS_STRIDE * 2;   // 263,168 B (<320KB WGP LDS)

    char* ws = (char*)d_ws;
    __bf16* Wbf = (__bf16*)ws;
    __bf16* Hbf = (__bf16*)(ws + wBytes);
    float* pmax = preconv ? (float*)(ws + wBytes + hBytes) : (float*)ws;
    float* psum    = pmax + (size_t)N_ROWS * SPLITS;
    float* ptot    = psum + (size_t)N_ROWS * SPLITS;
    float* rowloss = ptot + (size_t)N_ROWS * SPLITS;

    if (preconv) {
        cvt_to_bf16<<<(int)(wElems / 8 / 256), 256, 0, stream>>>(W, Wbf, (int)(wElems / 8));
        cvt_to_bf16<<<(int)(hElems / 8 / 256), 256, 0, stream>>>(H, Hbf, (int)(hElems / 8));
        (void)hipFuncSetAttribute(reinterpret_cast<const void*>(ce_gemm_stats<true>),
                                  hipFuncAttributeMaxDynamicSharedMemorySize, ldsBytes);
        ce_gemm_stats<true><<<dim3(N_ROWS / MTILE, SPLITS), 256, ldsBytes, stream>>>(
            Hbf, Wbf, pmax, psum, ptot);
    } else {
        (void)hipFuncSetAttribute(reinterpret_cast<const void*>(ce_gemm_stats<false>),
                                  hipFuncAttributeMaxDynamicSharedMemorySize, ldsBytes);
        ce_gemm_stats<false><<<dim3(N_ROWS / MTILE, SPLITS), 256, ldsBytes, stream>>>(
            H, W, pmax, psum, ptot);
    }
    ce_row_loss<<<N_ROWS / 8, 256, 0, stream>>>(
        H, W, labels, pmax, psum, ptot, rowloss);
    ce_reduce<<<1, 256, 0, stream>>>(rowloss, out);
}
